// ResNet50Conv5ROIFeatureExtractor_6038724018973
// MI455X (gfx1250) — matrix operand order repaired and checked by reading, not compiled, not run on hardware
//
#include <hip/hip_runtime.h>

// ---------------------------------------------------------------------------
// ResNet50 conv5 ROI feature extractor for MI455X (gfx1250).
// ~188 GFLOP of conv vs <0.3 GB of HBM traffic -> compute bound. Everything
// runs through v_wmma_f32_16x16x32_bf16 (f32 accumulate). Activations kept in
// K-major [C][N*H*W] bf16 layout so 1x1 convs are plain GEMMs and 3x3 convs
// are 9 shifted GEMMs accumulated in registers.
// Pipeline: double-buffered LDS, one barrier per K-slab; A-tile via CDNA5
// async global->LDS DMA (ASYNCcnt), B-tile register-staged + transposed.
// ---------------------------------------------------------------------------

typedef __bf16 bf16_t;
typedef __attribute__((ext_vector_type(16))) __bf16 bf16x16;
typedef __attribute__((ext_vector_type(8)))  __bf16 bf16x8;
typedef __attribute__((ext_vector_type(8)))  float  f32x8;

#define NCOLS   6272    // 128 rois * 7 * 7
#define NCOLS14 25088   // 128 rois * 14 * 14
#define FH 50
#define FW 76

#if defined(__has_builtin)
#  if __has_builtin(__builtin_amdgcn_global_load_async_to_lds_b128) && \
      __has_builtin(__builtin_amdgcn_s_wait_asynccnt)
#    define USE_ASYNC 1
#  else
#    define USE_ASYNC 0
#  endif
#else
#  define USE_ASYNC 0
#endif

#if USE_ASYNC
// Builtin signature (from clang diagnostic): int4 in AS1 (global) / AS3 (LDS).
typedef int v4i __attribute__((vector_size(16)));
typedef __attribute__((address_space(1))) v4i* g_v4i;
typedef __attribute__((address_space(3))) v4i* l_v4i;
#endif

// ------------------------- weight conversion -------------------------------
__global__ void cvt_w_1x1(const float* __restrict__ src, bf16_t* __restrict__ dst, int n) {
  int i = blockIdx.x * blockDim.x + threadIdx.x;
  if (i < n) dst[i] = (bf16_t)src[i];
}

// OIHW (3x3) -> [tap][O][I] bf16
__global__ void cvt_w_3x3(const float* __restrict__ src, bf16_t* __restrict__ dst, int OI) {
  int i = blockIdx.x * blockDim.x + threadIdx.x;
  if (i < 9 * OI) {
    int t  = i / OI;
    int oi = i - t * OI;
    dst[i] = (bf16_t)src[(size_t)oi * 9 + t];
  }
}

// ------------------------------ ROI align ----------------------------------
// grid = (128 rois, 8 channel groups of 128), block = 256
// pooled[c][roi*196 + ph*14 + pw]  (bf16, K-major for the GEMM stage)
__global__ __launch_bounds__(256)
void roi_align_pool(const float* __restrict__ feats, const float* __restrict__ rois,
                    const int* __restrict__ bidx, bf16_t* __restrict__ pooled) {
  const int n   = blockIdx.x;
  const int cg  = blockIdx.y;
  const int tid = threadIdx.x;

  __shared__ int   sy0[28], sy1[28], sx0[28], sx1[28];
  __shared__ float sly[28], slx[28], svy[28], svx[28];

  const float x1 = rois[n * 4 + 0] * 0.0625f;
  const float y1 = rois[n * 4 + 1] * 0.0625f;
  const float x2 = rois[n * 4 + 2] * 0.0625f;
  const float y2 = rois[n * 4 + 3] * 0.0625f;
  const float rw = fmaxf(x2 - x1, 1.0f);
  const float rh = fmaxf(y2 - y1, 1.0f);
  const int   b  = bidx[n];

  if (tid < 28) {
    const float g = ((float)tid + 0.5f) * 0.5f;
    float ys = y1 + g * (rh * (1.0f / 14.0f));
    svy[tid] = (ys > -1.0f && ys < (float)FH) ? 1.0f : 0.0f;
    ys = fminf(fmaxf(ys, 0.0f), (float)(FH - 1));
    float y0f = floorf(ys);
    int   y0  = (int)y0f;
    sy0[tid] = y0;
    sy1[tid] = (y0 + 1 < FH) ? y0 + 1 : FH - 1;
    sly[tid] = ys - y0f;

    float xs = x1 + g * (rw * (1.0f / 14.0f));
    svx[tid] = (xs > -1.0f && xs < (float)FW) ? 1.0f : 0.0f;
    xs = fminf(fmaxf(xs, 0.0f), (float)(FW - 1));
    float x0f = floorf(xs);
    int   x0  = (int)x0f;
    sx0[tid] = x0;
    sx1[tid] = (x0 + 1 < FW) ? x0 + 1 : FW - 1;
    slx[tid] = xs - x0f;
  }
  __syncthreads();

  const float* fb = feats + (size_t)b * 1024 * FH * FW;

  for (int e = tid; e < 128 * 196; e += 256) {
    const int c  = cg * 128 + (e / 196);
    const int p  = e - (e / 196) * 196;
    const int ph = p / 14, pw = p - (p / 14) * 14;
    const float* fc = fb + (size_t)c * (FH * FW);
    float sum = 0.0f;
#pragma unroll
    for (int i = 0; i < 2; i++) {
      const int   yi = ph * 2 + i;
      const int   y0 = sy0[yi], y1i = sy1[yi];
      const float ly = sly[yi], hy = 1.0f - ly, vy = svy[yi];
#pragma unroll
      for (int j = 0; j < 2; j++) {
        const int   xi = pw * 2 + j;
        const int   x0 = sx0[xi], x1c = sx1[xi];
        const float lx = slx[xi], hx = 1.0f - lx, vx = svx[xi];
        const float v00 = fc[y0 * FW + x0],  v01 = fc[y0 * FW + x1c];
        const float v10 = fc[y1i * FW + x0], v11 = fc[y1i * FW + x1c];
        sum += (v00 * hy * hx + v01 * hy * lx + v10 * ly * hx + v11 * ly * lx) * (vy * vx);
      }
    }
    pooled[(size_t)c * NCOLS14 + n * 196 + p] = (bf16_t)(sum * 0.25f);
  }
}

// ------------------------------ WMMA GEMM ----------------------------------
// D[M][N] = sum_taps A_t[M][K] * B_t[K][N], N = 6272 (49 tiles of 128).
// MODE: 0 = 1x1 identity columns, 1 = 1x1 stride-2 from 14x14, 2 = 3x3 9-tap.
// EPI : 0 = scale/bias + relu -> bf16
//       1 = scale/bias (no relu) -> bf16            (downsample residual)
//       2 = scale/bias + residual + relu -> bf16
//       3 = scale/bias + residual + relu -> f32 NCHW (final output)
template<int MODE, int EPI>
__global__ __launch_bounds__(256)
void gemm_bf16_wmma(const bf16_t* __restrict__ A, const bf16_t* __restrict__ B,
                    const float* __restrict__ scale, const float* __restrict__ bias,
                    const bf16_t* __restrict__ resid,
                    bf16_t* __restrict__ outB, float* __restrict__ outF,
                    int M, int K) {
  __shared__ __align__(16) bf16_t As[2][128][40];  // row-major [m][k], padded
  __shared__ __align__(16) bf16_t Bs[2][128][40];  // col-major [n][k], padded

  const int tid   = threadIdx.x;
  const int lane  = tid & 31;
  const int lh    = lane >> 4;
  const int lm    = lane & 15;
  const int wave  = tid >> 5;
  const int wm    = wave & 3;    // 4 waves along M (32 rows each)
  const int wn    = wave >> 2;   // 2 waves along N (64 cols each)
  const int nBase = blockIdx.x * 128;
  const int mBase = blockIdx.y * 128;

  const int kslabs = K >> 5;
  const int ntaps  = (MODE == 2) ? 9 : 1;
  const int total  = ntaps * kslabs;

  // ---- per-thread load geometry (K-loop invariant, hoisted once) ----
  const int ar = tid >> 1;            // A row 0..127
  const int ah = (tid & 1) << 4;      // A k-half 0/16
  const bf16_t* aRow = A + (size_t)(mBase + ar) * K + ah;

  const int nn  = tid & 127;          // B column handled by this thread
  const int kst = tid >> 7;           // starting k parity (0/1), step 2
  const int n    = nBase + nn;
  const int roiN = n / 49;
  const int remN = n - roiN * 49;
  const int oyN  = remN / 7;
  const int oxN  = remN - oyN * 7;
  const int inCol1 = roiN * 196 + oyN * 28 + oxN * 2;   // stride-2 source col

  f32x8  acc[2][4] = {};
  bf16x8 aStage0, aStage1;
  bf16x8 bStageV0, bStageV1;
  bf16_t bStageS[16];

  // issue global loads for slab (A async straight into LDS buf when available)
  auto issueA = [&](int slab, int buf) {
    const int t  = (MODE == 2) ? (slab / kslabs) : 0;
    const int ks = slab - t * kslabs;
    const bf16_t* src = aRow + (size_t)t * M * K + (ks << 5);
#if USE_ASYNC
    bf16_t* dst = &As[buf][ar][ah];
    __builtin_amdgcn_global_load_async_to_lds_b128((g_v4i)src, (l_v4i)dst, 0, 0);
    __builtin_amdgcn_global_load_async_to_lds_b128((g_v4i)(src + 8), (l_v4i)(dst + 8), 0, 0);
#else
    (void)buf;
    aStage0 = *(const bf16x8*)src;
    aStage1 = *(const bf16x8*)(src + 8);
#endif
  };
  auto issueB = [&](int slab) {
    const int t  = (MODE == 2) ? (slab / kslabs) : 0;
    const int ks = slab - t * kslabs;
    const int k0 = ks << 5;
    if (MODE == 0) {
      const bf16_t* src = B + (size_t)(k0 + (tid >> 4)) * NCOLS + nBase + ((tid & 15) << 3);
      bStageV0 = *(const bf16x8*)src;
      bStageV1 = *(const bf16x8*)(src + (size_t)16 * NCOLS);
    } else if (MODE == 1) {
      const bf16_t* src = B + (size_t)(k0 + kst) * NCOLS14 + inCol1;
#pragma unroll
      for (int i = 0; i < 16; i++) { bStageS[i] = *src; src += 2 * NCOLS14; }
    } else {
      const int ty = t / 3, tx = t - (t / 3) * 3;
      const int iy = oyN + ty - 1, ix = oxN + tx - 1;
      const bool ok = ((unsigned)iy < 7u) && ((unsigned)ix < 7u);
      const bf16_t* src = B + (size_t)(k0 + kst) * NCOLS + roiN * 49 + iy * 7 + ix;
#pragma unroll
      for (int i = 0; i < 16; i++) {
        bStageS[i] = ok ? *src : (bf16_t)0.0f;
        src += 2 * NCOLS;
      }
    }
  };
  // write staged regs into LDS buffer `buf`
  auto commit = [&](int buf) {
#if !USE_ASYNC
    *(bf16x8*)(&As[buf][ar][ah])     = aStage0;
    *(bf16x8*)(&As[buf][ar][ah + 8]) = aStage1;
#endif
    if (MODE == 0) {
      const int k  = tid >> 4;
      const int n8 = (tid & 15) << 3;
#pragma unroll
      for (int j = 0; j < 8; j++) {
        Bs[buf][n8 + j][k]      = bStageV0[j];
        Bs[buf][n8 + j][k + 16] = bStageV1[j];
      }
    } else {
#pragma unroll
      for (int i = 0; i < 16; i++) Bs[buf][nn][kst + 2 * i] = bStageS[i];
    }
  };

  issueA(0, 0);
  issueB(0);

  for (int s = 0; s < total; ++s) {
    const int cur = s & 1;
    commit(cur);                          // tile s: staged regs -> LDS
#if USE_ASYNC
    __builtin_amdgcn_s_wait_asynccnt(0);  // tile s A-DMA arrived
#endif
    __syncthreads();                      // tile s visible to all waves
    if (s + 1 < total) {                  // overlap next tile's loads w/ WMMA
      issueA(s + 1, cur ^ 1);
      issueB(s + 1);
    }
    // ---- fragments (ISA 7.12.2 striped layouts) + WMMA ----
    bf16x16 afr[2], bfr[4];
#pragma unroll
    for (int mi = 0; mi < 2; mi++) {
      const bf16_t* p = &As[cur][wm * 32 + mi * 16 + lm][lh * 8];
      bf16x8 lo = *(const bf16x8*)(p);       // K = base .. base+7
      bf16x8 hi = *(const bf16x8*)(p + 16);  // K = base+16 .. base+23
      afr[mi] = __builtin_shufflevector(lo, hi,
                0,1,2,3,4,5,6,7,8,9,10,11,12,13,14,15);
    }
#pragma unroll
    for (int ni = 0; ni < 4; ni++) {
      const bf16_t* q = &Bs[cur][wn * 64 + ni * 16 + lm][lh * 8];
      bf16x8 blo = *(const bf16x8*)(q);
      bf16x8 bhi = *(const bf16x8*)(q + 16);
      bfr[ni] = __builtin_shufflevector(blo, bhi,
                0,1,2,3,4,5,6,7,8,9,10,11,12,13,14,15);
    }
#pragma unroll
    for (int mi = 0; mi < 2; mi++)
#pragma unroll
      for (int ni = 0; ni < 4; ni++)
        acc[mi][ni] = __builtin_amdgcn_wmma_f32_16x16x32_bf16(
            false, afr[mi], false, bfr[ni], (short)0, acc[mi][ni], false, false);
  }

  // ---- epilogue: BN scale/bias (+residual) (+relu), bf16 and/or f32 NCHW ---
#pragma unroll
  for (int mi = 0; mi < 2; mi++) {
#pragma unroll
    for (int r = 0; r < 8; r++) {
      const int   m  = mBase + wm * 32 + mi * 16 + r + lh * 8;
      const float sc = scale[m];
      const float bi = bias[m];
#pragma unroll
      for (int ni = 0; ni < 4; ni++) {
        const int nc = nBase + wn * 64 + ni * 16 + lm;
        float v = acc[mi][ni][r] * sc + bi;
        if (EPI >= 2) v += (float)resid[(size_t)m * NCOLS + nc];
        if (EPI != 1) v = fmaxf(v, 0.0f);
        if (EPI == 3) {
          const int roi = nc / 49;
          const int rem = nc - roi * 49;
          outF[((size_t)roi * 2048 + m) * 49 + rem] = v;
        } else {
          outB[(size_t)m * NCOLS + nc] = (bf16_t)v;
        }
      }
    }
  }
}

// ------------------------------ launcher -----------------------------------
extern "C" void kernel_launch(void* const* d_in, const int* in_sizes, int n_in,
                              void* d_out, int out_size, void* d_ws, size_t ws_size,
                              hipStream_t stream) {
  (void)in_sizes; (void)n_in; (void)out_size; (void)ws_size;
  const float* feats = (const float*)d_in[0];
  const float* rois  = (const float*)d_in[1];
  const int*   bidx  = (const int*)d_in[2];

  // params pytree flatten: list of dicts, dict keys sorted alphabetically.
  const float *b1_0=(const float*)d_in[3],  *b2_0=(const float*)d_in[4],
              *b3_0=(const float*)d_in[5],  *bd_0=(const float*)d_in[6],
              *s1_0=(const float*)d_in[7],  *s2_0=(const float*)d_in[8],
              *s3_0=(const float*)d_in[9],  *sd_0=(const float*)d_in[10],
              *w1_0=(const float*)d_in[11], *w2_0=(const float*)d_in[12],
              *w3_0=(const float*)d_in[13], *wd_0=(const float*)d_in[14];
  const float *b1_1=(const float*)d_in[15], *b2_1=(const float*)d_in[16],
              *b3_1=(const float*)d_in[17], *s1_1=(const float*)d_in[18],
              *s2_1=(const float*)d_in[19], *s3_1=(const float*)d_in[20],
              *w1_1=(const float*)d_in[21], *w2_1=(const float*)d_in[22],
              *w3_1=(const float*)d_in[23];
  const float *b1_2=(const float*)d_in[24], *b2_2=(const float*)d_in[25],
              *b3_2=(const float*)d_in[26], *s1_2=(const float*)d_in[27],
              *s2_2=(const float*)d_in[28], *s3_2=(const float*)d_in[29],
              *w1_2=(const float*)d_in[30], *w2_2=(const float*)d_in[31],
              *w3_2=(const float*)d_in[32];

  // ---- workspace carve-up (256B aligned) ----
  char* ws = (char*)d_ws;
  size_t off = 0;
  auto take = [&](size_t bytes) {
    void* p = ws + off;
    off = (off + bytes + 255) & ~(size_t)255;
    return p;
  };
  bf16_t* pooled = (bf16_t*)take((size_t)1024 * NCOLS14 * 2);
  bf16_t* wb0_1  = (bf16_t*)take((size_t)512 * 1024 * 2);
  bf16_t* wb0_2  = (bf16_t*)take((size_t)9 * 512 * 512 * 2);
  bf16_t* wb0_3  = (bf16_t*)take((size_t)2048 * 512 * 2);
  bf16_t* wb0_d  = (bf16_t*)take((size_t)2048 * 1024 * 2);
  bf16_t* wb1_1  = (bf16_t*)take((size_t)512 * 2048 * 2);
  bf16_t* wb1_2  = (bf16_t*)take((size_t)9 * 512 * 512 * 2);
  bf16_t* wb1_3  = (bf16_t*)take((size_t)2048 * 512 * 2);
  bf16_t* wb2_1  = (bf16_t*)take((size_t)512 * 2048 * 2);
  bf16_t* wb2_2  = (bf16_t*)take((size_t)9 * 512 * 512 * 2);
  bf16_t* wb2_3  = (bf16_t*)take((size_t)2048 * 512 * 2);
  bf16_t* mid    = (bf16_t*)take((size_t)512 * NCOLS * 2);
  bf16_t* mid2   = (bf16_t*)take((size_t)512 * NCOLS * 2);
  bf16_t* resid  = (bf16_t*)take((size_t)2048 * NCOLS * 2);
  bf16_t* actA   = pooled;  // alias: pooled dead after blk0 conv1+downsample
  bf16_t* actB   = resid;   // alias: downsample residual dead after blk0 conv3

  // ---- 1) weight conversion ----
  auto c1 = [&](const float* s, bf16_t* d, int n) {
    cvt_w_1x1<<<(n + 255) / 256, 256, 0, stream>>>(s, d, n);
  };
  auto c3 = [&](const float* s, bf16_t* d, int OI) {
    cvt_w_3x3<<<(9 * OI + 255) / 256, 256, 0, stream>>>(s, d, OI);
  };
  c1(w1_0, wb0_1, 512 * 1024);  c3(w2_0, wb0_2, 512 * 512);
  c1(w3_0, wb0_3, 2048 * 512);  c1(wd_0, wb0_d, 2048 * 1024);
  c1(w1_1, wb1_1, 512 * 2048);  c3(w2_1, wb1_2, 512 * 512);
  c1(w3_1, wb1_3, 2048 * 512);
  c1(w1_2, wb2_1, 512 * 2048);  c3(w2_2, wb2_2, 512 * 512);
  c1(w3_2, wb2_3, 2048 * 512);

  // ---- 2) ROI align -> pooled bf16 [1024][25088] ----
  roi_align_pool<<<dim3(128, 8), 256, 0, stream>>>(feats, rois, bidx, pooled);

  // ---- 3) bottlenecks as WMMA GEMMs ----
  dim3 g2048(49, 16), g512(49, 4);
  // block 0
  gemm_bf16_wmma<1,1><<<g2048, 256, 0, stream>>>(wb0_d, pooled, sd_0, bd_0, nullptr, resid, nullptr, 2048, 1024);
  gemm_bf16_wmma<1,0><<<g512,  256, 0, stream>>>(wb0_1, pooled, s1_0, b1_0, nullptr, mid,   nullptr,  512, 1024);
  gemm_bf16_wmma<2,0><<<g512,  256, 0, stream>>>(wb0_2, mid,    s2_0, b2_0, nullptr, mid2,  nullptr,  512,  512);
  gemm_bf16_wmma<0,2><<<g2048, 256, 0, stream>>>(wb0_3, mid2,   s3_0, b3_0, resid,   actA,  nullptr, 2048,  512);
  // block 1
  gemm_bf16_wmma<0,0><<<g512,  256, 0, stream>>>(wb1_1, actA,   s1_1, b1_1, nullptr, mid,   nullptr,  512, 2048);
  gemm_bf16_wmma<2,0><<<g512,  256, 0, stream>>>(wb1_2, mid,    s2_1, b2_1, nullptr, mid2,  nullptr,  512,  512);
  gemm_bf16_wmma<0,2><<<g2048, 256, 0, stream>>>(wb1_3, mid2,   s3_1, b3_1, actA,    actB,  nullptr, 2048,  512);
  // block 2 (final conv writes f32 NCHW straight to d_out)
  gemm_bf16_wmma<0,0><<<g512,  256, 0, stream>>>(wb2_1, actB,   s1_2, b1_2, nullptr, mid,   nullptr,  512, 2048);
  gemm_bf16_wmma<2,0><<<g512,  256, 0, stream>>>(wb2_2, mid,    s2_2, b2_2, nullptr, mid2,  nullptr,  512,  512);
  gemm_bf16_wmma<0,3><<<g2048, 256, 0, stream>>>(wb2_3, mid2,   s3_2, b3_2, actB, nullptr, (float*)d_out, 2048, 512);
}
